// JTMessPassNet_67207648248163
// MI455X (gfx1250) — compile-verified, matrix operand order checked by reading, not run on hardware
//
#include <hip/hip_runtime.h>
#include <hip/hip_bf16.h>

// ---- problem constants (match reference) ----
#define ATOM_FDIM 35
#define BOND_FDIM 5
#define BOND_IN   (ATOM_FDIM + BOND_FDIM)   // 40
#define HIDDEN    128
#define HID_PK    (HIDDEN / 2)              // 64 packed bf16 pairs per row
#define MAX_NB    10
#define A_TOT     131072
#define B_TOT     262144
#define M_TOT     32768
#define N_MOLS_C  4096
#define DEPTH     6

typedef __attribute__((ext_vector_type(16))) __bf16   v16bf;
typedef __attribute__((ext_vector_type(2)))  __bf16   v2bf;
typedef __attribute__((ext_vector_type(8)))  float    v8f;
typedef __attribute__((ext_vector_type(4)))  unsigned v4u;
typedef __attribute__((ext_vector_type(8)))  unsigned v8u;

#define NEI_STR 132   // LDS row stride (128 + 4 pad) -> conflict-free banks

// -----------------------------------------------------------------------------
// Pack W_h (128x128) and W_o[:,35:163] (128x128) into the wave32 bf16 B-matrix
// fragment layout (ISA 7.12.2, 16-bit 32x16 operand):
//   frag[nt][ks][lane][e]: lane<16 -> col nt*16+lane, K = ks*32 + {0..7,16..23}
//                          lane>=16 -> col nt*16+lane-16, K = ks*32 + {8..15,24..31}
// -----------------------------------------------------------------------------
__global__ void prep_frags(const float* __restrict__ Wh, const float* __restrict__ Wo,
                           __bf16* __restrict__ whf, __bf16* __restrict__ wof) {
    int tid = blockIdx.x * blockDim.x + threadIdx.x;   // 0..32767
    int which = tid >> 14;                              // 0 = Wh, 1 = Wo(K part)
    int t    = tid & 16383;
    int e    = t & 15;
    int lane = (t >> 4) & 31;
    int ks   = (t >> 9) & 3;
    int nt   = t >> 11;
    int g    = nt * 16 + (lane & 15);
    int kb   = (lane & 16) >> 1;                        // 0 or 8
    int k    = ks * 32 + ((e < 8) ? (kb + e) : (16 + kb + (e - 8)));
    float v  = (which == 0) ? Wh[g * HIDDEN + k]
                            : Wo[g * (ATOM_FDIM + HIDDEN) + ATOM_FDIM + k];
    (which == 0 ? whf : wof)[t] = (__bf16)v;
}

// Convert tree_message [M,128] f32 -> packed bf16 once (shrinks gather rows).
__global__ void cvt_tree(const float* __restrict__ tree, __bf16* __restrict__ treeb) {
    const size_t i = (size_t)blockIdx.x * blockDim.x + threadIdx.x; // M*64 threads
    const float2 f = ((const float2*)tree)[i];
    v2bf p; p[0] = (__bf16)f.x; p[1] = (__bf16)f.y;
    ((v2bf*)treeb)[i] = p;
}

// -----------------------------------------------------------------------------
// h0 = ReLU(bond_feats @ W_i^T), K=40: tiny FLOPs -> VALU.
// Writes h0 (f32, reused every iteration) and seeds graph_message (bf16).
// -----------------------------------------------------------------------------
__global__ void h0_kernel(const float* __restrict__ bf, const float* __restrict__ Wi,
                          float* __restrict__ h0, __bf16* __restrict__ gmA) {
    __shared__ float lbf[2][BOND_IN];
    const int b0 = blockIdx.x * 2;
    const int t  = threadIdx.x;
    if (t < 2 * BOND_IN) lbf[t / BOND_IN][t % BOND_IN] = bf[(size_t)b0 * BOND_IN + t];
    __syncthreads();
    const int bl = t >> 7;
    const int h  = t & 127;
    const float* w = Wi + h * BOND_IN;
    const float* x = lbf[bl];
    float acc = 0.f;
    #pragma unroll
    for (int f = 0; f < BOND_IN; ++f) acc = fmaf(x[f], w[f], acc);
    acc = fmaxf(acc, 0.f);
    const size_t o = (size_t)(b0 + bl) * HIDDEN + h;
    h0[o]  = acc;
    gmA[o] = (__bf16)acc;
}

// -----------------------------------------------------------------------------
// One message-passing iteration. Block = 64 bonds (4 M-tiles), 8 waves.
// Wave 0 issues a TDM tensor_load_to_lds for the dense 64x128 f32 h0 C-init
// tile (32KB contiguous), with D# LDS-padding of 4 DWORDs every 128 DWORDs so
// the tile lands with the same 132-DWORD stride as nei. The DMA overlaps the
// random gather phase. Then 16x v_wmma_f32_16x16x32_bf16 per wave with W_h
// B-fragments cached in registers; ReLU; store new graph_message as bf16.
// -----------------------------------------------------------------------------
__global__ void mp_iter(const unsigned* __restrict__ treeb,  // [M,64] packed bf16
                        const unsigned* __restrict__ gm_in,  // [B,64] packed bf16
                        const int*   __restrict__ bond_graph,// [B,10]
                        const float* __restrict__ h0,        // [B,128] f32
                        const __bf16* __restrict__ whf,      // packed W_h frags
                        __bf16* __restrict__ gm_out) {       // [B,128] bf16
    __shared__ float nei[64 * NEI_STR];
    __shared__ float h0t[64 * NEI_STR];   // TDM destination (padded rows)
    __shared__ int   idx[64 * MAX_NB];
    const int b0   = blockIdx.x * 64;
    const int t    = threadIdx.x;
    const int wave = t >> 5;

    // ---- wave 0: kick off TDM load of the h0 tile (async, TENSORcnt) ----
    if (wave == 0) {
        const unsigned long long ga = (unsigned long long)(uintptr_t)(h0 + (size_t)b0 * HIDDEN);
        const unsigned lds = (unsigned)(uintptr_t)&h0t[0];   // low 32 bits = LDS offset
        v4u g0;
        g0[0] = 1u;                                          // count=1 (valid user D#)
        g0[1] = lds;                                         // lds_addr
        g0[2] = (unsigned)ga;                                // global_addr[31:0]
        g0[3] = (unsigned)((ga >> 32) & 0x01FFFFFFu) | (2u << 30); // ga[56:32] | type=2
        v8u g1;
        // data_size=4B, pad_enable, pad_interval=128 DW, pad_amount=4 DW
        g1[0] = (2u << 16) | (1u << 20) | (6u << 22) | (3u << 25);
        g1[1] = (8192u & 0xFFFFu) << 16;                     // tensor_dim0 = 8192 (lo16)
        g1[2] = (1u << 16);                                  // tensor_dim0 hi=0 | tensor_dim1=1
        g1[3] = (8192u << 16);                               // tensor_dim1 hi=0 | tile_dim0=8192
        g1[4] = 1u;                                          // tile_dim1=1, tile_dim2=0
        g1[5] = 8192u;                                       // tensor_dim0_stride lo32
        g1[6] = 0u;                                          // stride0 hi | stride1 lo
        g1[7] = 0u;                                          // tensor_dim1_stride hi
        asm volatile("tensor_load_to_lds %0, %1" : : "s"(g0), "s"(g1) : "memory");
    }

    for (int p = t; p < 64 * MAX_NB; p += 256)
        idx[p] = bond_graph[(size_t)b0 * MAX_NB + p];
    __syncthreads();

    // ---- gather-sum (overlaps the TDM): 64 rows x 64 packed bf16 pairs ----
    #pragma unroll
    for (int p = t; p < 64 * HID_PK; p += 256) {
        const int bl = p >> 6, hp = p & 63;
        float s0 = 0.f, s1 = 0.f;
        #pragma unroll
        for (int j = 0; j < MAX_NB; ++j) {
            const int ix = idx[bl * MAX_NB + j];
            const unsigned* row = (ix < M_TOT) ? (treeb + (size_t)ix * HID_PK)
                                               : (gm_in + (size_t)(ix - M_TOT) * HID_PK);
            const unsigned u = row[hp];
            s0 += __uint_as_float(u << 16);
            s1 += __uint_as_float(u & 0xffff0000u);
        }
        nei[bl * NEI_STR + 2 * hp]     = s0;
        nei[bl * NEI_STR + 2 * hp + 1] = s1;
    }
    if (wave == 0) __builtin_amdgcn_s_wait_tensorcnt(0);  // h0 tile landed in LDS
    __syncthreads();                                      // publish to all waves

    const int lane = t & 31;
    const int nt   = wave;
    const int row  = lane & 15;              // A-matrix M index / C column index
    const int kb   = (lane & 16) >> 1;       // 0 or 8 (A K-offset, C row-base)
    const int g    = nt * 16 + row;          // output column

    const v16bf* wv = (const v16bf*)whf;
    v16bf bfr[4];
    #pragma unroll
    for (int ks = 0; ks < 4; ++ks) bfr[ks] = wv[(nt * 4 + ks) * 32 + lane];

    #pragma unroll
    for (int mt = 0; mt < 4; ++mt) {
        const int r0 = mt * 16 + kb;         // row within block tile
        v8f acc;
        #pragma unroll
        for (int r = 0; r < 8; ++r)
            acc[r] = h0t[(r0 + r) * NEI_STR + g];
        #pragma unroll
        for (int ks = 0; ks < 4; ++ks) {
            v16bf a;
            const float* nr = &nei[(mt * 16 + row) * NEI_STR + ks * 32 + kb];
            #pragma unroll
            for (int e = 0; e < 8; ++e) { a[e] = (__bf16)nr[e]; a[8 + e] = (__bf16)nr[16 + e]; }
            acc = __builtin_amdgcn_wmma_f32_16x16x32_bf16(false, a, false, bfr[ks],
                                                          (short)0, acc, false, false);
        }
        #pragma unroll
        for (int r = 0; r < 8; ++r)
            gm_out[(size_t)(b0 + r0 + r) * HIDDEN + g] = (__bf16)fmaxf(acc[r], 0.f);
    }
}

// -----------------------------------------------------------------------------
// Atom readout + per-molecule mean. Block = 1 molecule (32 atoms).
// K=35 atom_feats part via VALU straight into the C accumulator (+bias);
// K=128 via 2x4 WMMAs (B-frags cached); ReLU; shfl-reduce -> mean.
// -----------------------------------------------------------------------------
__global__ void atom_kernel(const unsigned* __restrict__ treeb,
                            const unsigned* __restrict__ gm,     // final graph_message
                            const int*   __restrict__ atom_graph,// [A,10]
                            const float* __restrict__ af,        // [A,35]
                            const float* __restrict__ Wo,        // [128,163]
                            const float* __restrict__ bo,        // [128]
                            const __bf16* __restrict__ wof,      // packed W_o[:,35:]
                            float* __restrict__ out) {           // [4096,128]
    __shared__ float nei[32 * NEI_STR];
    __shared__ float afs[32][ATOM_FDIM + 1];
    __shared__ int   idx[32 * MAX_NB];
    const int m  = blockIdx.x;
    const int a0 = m * 32;
    const int t  = threadIdx.x;
    for (int p = t; p < 32 * MAX_NB; p += 256)
        idx[p] = atom_graph[(size_t)a0 * MAX_NB + p];
    for (int p = t; p < 32 * ATOM_FDIM; p += 256)
        afs[p / ATOM_FDIM][p % ATOM_FDIM] = af[(size_t)a0 * ATOM_FDIM + p];
    __syncthreads();

    #pragma unroll
    for (int p = t; p < 32 * HID_PK; p += 256) {
        const int al = p >> 6, hp = p & 63;
        float s0 = 0.f, s1 = 0.f;
        #pragma unroll
        for (int j = 0; j < MAX_NB; ++j) {
            const int ix = idx[al * MAX_NB + j];
            const unsigned* row = (ix < M_TOT) ? (treeb + (size_t)ix * HID_PK)
                                               : (gm + (size_t)(ix - M_TOT) * HID_PK);
            const unsigned u = row[hp];
            s0 += __uint_as_float(u << 16);
            s1 += __uint_as_float(u & 0xffff0000u);
        }
        nei[al * NEI_STR + 2 * hp]     = s0;
        nei[al * NEI_STR + 2 * hp + 1] = s1;
    }
    __syncthreads();

    const int lane = t & 31;
    const int nt   = t >> 5;
    const int col  = lane & 15;
    const int g    = nt * 16 + col;
    const int kb   = (lane & 16) >> 1;

    float wrow[ATOM_FDIM];
    #pragma unroll
    for (int f = 0; f < ATOM_FDIM; ++f) wrow[f] = Wo[g * (ATOM_FDIM + HIDDEN) + f];
    const float bias = bo[g];

    const v16bf* wv = (const v16bf*)wof;
    v16bf bfr[4];
    #pragma unroll
    for (int ks = 0; ks < 4; ++ks) bfr[ks] = wv[(nt * 4 + ks) * 32 + lane];

    float total = 0.f;
    #pragma unroll
    for (int mt = 0; mt < 2; ++mt) {
        v8f acc;
        #pragma unroll
        for (int r = 0; r < 8; ++r) {
            const int arow = mt * 16 + kb + r;
            float c = bias;
            #pragma unroll
            for (int f = 0; f < ATOM_FDIM; ++f) c = fmaf(afs[arow][f], wrow[f], c);
            acc[r] = c;
        }
        #pragma unroll
        for (int ks = 0; ks < 4; ++ks) {
            v16bf a;
            const float* nr = &nei[(mt * 16 + col) * NEI_STR + ks * 32 + kb];
            #pragma unroll
            for (int e = 0; e < 8; ++e) { a[e] = (__bf16)nr[e]; a[8 + e] = (__bf16)nr[16 + e]; }
            acc = __builtin_amdgcn_wmma_f32_16x16x32_bf16(false, a, false, bfr[ks],
                                                          (short)0, acc, false, false);
        }
        float s = 0.f;
        #pragma unroll
        for (int r = 0; r < 8; ++r) s += fmaxf(acc[r], 0.f);
        total += s;
    }
    total += __shfl_xor(total, 16, 32);   // lanes L and L+16 hold complementary rows
    if (lane < 16) out[(size_t)m * HIDDEN + g] = total * (1.0f / 32.0f);
}

// -----------------------------------------------------------------------------
extern "C" void kernel_launch(void* const* d_in, const int* in_sizes, int n_in,
                              void* d_out, int out_size, void* d_ws, size_t ws_size,
                              hipStream_t stream) {
    const float* atom_feats = (const float*)d_in[0];
    const float* bond_feats = (const float*)d_in[1];
    const float* tree       = (const float*)d_in[2];
    const int*   atom_graph = (const int*)  d_in[3];
    const int*   bond_graph = (const int*)  d_in[4];
    // d_in[5] = n_mols (compile-time 4096)
    const float* Wi = (const float*)d_in[6];
    const float* Wh = (const float*)d_in[7];
    const float* Wo = (const float*)d_in[8];
    const float* bo = (const float*)d_in[9];

    // workspace layout (bf16 ping-pong keeps the gather set ~L2-resident)
    float*  h0    = (float*)d_ws;                              // B*128 f32
    __bf16* gmA   = (__bf16*)(h0 + (size_t)B_TOT * HIDDEN);    // B*128 bf16
    __bf16* gmB   = gmA + (size_t)B_TOT * HIDDEN;              // B*128 bf16
    __bf16* treeb = gmB + (size_t)B_TOT * HIDDEN;              // M*128 bf16
    __bf16* whf   = treeb + (size_t)M_TOT * HIDDEN;            // 16384 bf16
    __bf16* wof   = whf + 16384;                               // 16384 bf16

    cvt_tree  <<<(M_TOT * HID_PK) / 256, 256, 0, stream>>>(tree, treeb);
    prep_frags<<<128, 256, 0, stream>>>(Wh, Wo, whf, wof);
    h0_kernel <<<B_TOT / 2, 256, 0, stream>>>(bond_feats, Wi, h0, gmA);

    __bf16* cur = gmA;
    __bf16* nxt = gmB;
    for (int d = 0; d < DEPTH - 1; ++d) {
        mp_iter<<<B_TOT / 64, 256, 0, stream>>>((const unsigned*)treeb, (const unsigned*)cur,
                                                bond_graph, h0, whf, nxt);
        __bf16* tmp = cur; cur = nxt; nxt = tmp;
    }
    atom_kernel<<<N_MOLS_C, 256, 0, stream>>>((const unsigned*)treeb, (const unsigned*)cur,
                                              atom_graph, atom_feats, Wo, bo, wof,
                                              (float*)d_out);
}
